// TransformerIrisWorldModel_20478404068002
// MI455X (gfx1250) — matrix operand order repaired and checked by reading, not compiled, not run on hardware
//
#include <hip/hip_runtime.h>
#include <hip/hip_bf16.h>

typedef __attribute__((ext_vector_type(16))) __bf16 v16bf;
typedef __attribute__((ext_vector_type(8)))  __bf16 v8bf;
typedef __attribute__((ext_vector_type(8)))  float  v8f;

#define BB      2
#define TSEQ    1024
#define DMODEL  1024
#define NHEAD   16
#define DHEAD   64
#define NLAYER  4
#define FFDIM   4096
#define ODIM    512
#define MROWS   (BB*TSEQ)

#define EP_NONE 0
#define EP_GELU 1
#define EP_RES  2

union FragA { v16bf v; v8bf h[2]; };

__device__ __forceinline__ float gelu_exact(float x) {
    return 0.5f * x * (1.0f + erff(x * 0.70710678118654752f));
}

// ---------------------------------------------------------------------------
// GEMM: C[M,N] = epilogue(A[M,K] @ W[K,N] + bias), A/W fp32 in global,
// converted to bf16 on the fly; fp32 WMMA accumulation.
// Block tile 128x256, 8 waves, each wave a 64x64 sub-tile (4x4 WMMA frags).
// Register-buffered + LDS double-buffered pipeline:
//   iter kt: [issue global loads of tile kt+1 into regs] ->
//            [ds_load frags + 16 WMMA on tile kt]        ->
//            [cvt + packed ds_store_b128 of tile kt+1]   -> barrier
// ---------------------------------------------------------------------------
__global__ __launch_bounds__(256)
void gemm_wmma_bf16(const float* __restrict__ A, const float* __restrict__ W,
                    const float* __restrict__ bias, const float* __restrict__ Res,
                    float* __restrict__ Cf, __bf16* __restrict__ Cb,
                    int M, int N, int K, int epilogue)
{
    __shared__ __bf16 Asl[2][128 * 32];   // [buf][row][k]
    __shared__ __bf16 Bsl[2][256 * 32];   // [buf][n][k]  (transposed stage)

    const int tid  = threadIdx.x;
    const int lane = tid & 31;
    const int wave = tid >> 5;
    const int wm   = wave >> 2;        // 0..1  -> m offset wm*64
    const int wn   = wave & 3;         // 0..3  -> n offset wn*64
    const int bm   = blockIdx.y * 128;
    const int bn   = blockIdx.x * 256;
    const int l15  = lane & 15;
    const int klo  = (lane < 16) ? 0 : 8;    // unused (A-frag K runs also start at ks)
    const int ks   = (lane < 16) ? 0 : 16;   // B-frag K run start

    // staging thread->data map
    const int ar   = tid >> 1;          // A row 0..127
    const int ah   = (tid & 1) * 16;    // A col half: 0 or 16
    // B: this thread owns column n = tid (0..255), all 32 k's

    float aReg[16];
    float bReg[32];

    auto loadRegs = [&](int k0) {
#pragma unroll
        for (int i = 0; i < 16; ++i)
            aReg[i] = A[(size_t)(bm + ar) * K + (k0 + ah + i)];
#pragma unroll
        for (int k = 0; k < 32; ++k)
            bReg[k] = W[(size_t)(k0 + k) * N + (bn + tid)];
    };
    auto storeRegs = [&](int buf) {
#pragma unroll
        for (int c = 0; c < 2; ++c) {
            v8bf ch;
#pragma unroll
            for (int e = 0; e < 8; ++e) ch[e] = (__bf16)aReg[c * 8 + e];
            *(v8bf*)&Asl[buf][ar * 32 + ah + c * 8] = ch;
        }
#pragma unroll
        for (int c = 0; c < 4; ++c) {
            v8bf ch;
#pragma unroll
            for (int e = 0; e < 8; ++e) ch[e] = (__bf16)bReg[c * 8 + e];
            *(v8bf*)&Bsl[buf][tid * 32 + c * 8] = ch;
        }
    };

    v8f zero = {};
    v8f acc[4][4];
#pragma unroll
    for (int mt = 0; mt < 4; ++mt)
#pragma unroll
        for (int nt = 0; nt < 4; ++nt) acc[mt][nt] = zero;

    const int nk = K >> 5;
    loadRegs(0);
    storeRegs(0);
    __syncthreads();

    for (int kt = 0; kt < nk; ++kt) {
        const int cur = kt & 1;

        // (1) issue next tile's global loads (results land in regs)
        if (kt + 1 < nk) loadRegs((kt + 1) * 32);

        // cache-prefetch the tile after next
        if (kt + 2 < nk) {
            const int kp = (kt + 2) * 32;
            int r = tid & 127, half = tid >> 7;
            __builtin_prefetch(&A[(size_t)(bm + r) * K + kp + half * 16], 0, 0);
#pragma unroll
            for (int q = 0; q < 2; ++q) {
                int idx = q * 256 + tid;
                int rr = idx >> 4, cl = idx & 15;
                __builtin_prefetch(&W[(size_t)(kp + rr) * N + bn + cl * 16], 0, 0);
            }
        }

        // (2) compute on current tile
        FragA fa[4];
#pragma unroll
        for (int mt = 0; mt < 4; ++mt) {
            int row = wm * 64 + mt * 16 + l15;
            fa[mt].h[0] = *(const v8bf*)&Asl[cur][row * 32 + ks];
            fa[mt].h[1] = *(const v8bf*)&Asl[cur][row * 32 + ks + 8];
        }
#pragma unroll
        for (int nt = 0; nt < 4; ++nt) {
            int col = wn * 64 + nt * 16 + l15;
            v16bf fb = *(const v16bf*)&Bsl[cur][col * 32 + ks];
#pragma unroll
            for (int mt = 0; mt < 4; ++mt) {
                acc[mt][nt] = __builtin_amdgcn_wmma_f32_16x16x32_bf16(
                    false, fa[mt].v, false, fb, (short)0, acc[mt][nt], false, false);
            }
        }

        // (3) convert + packed LDS stores of next tile
        if (kt + 1 < nk) storeRegs(cur ^ 1);
        __syncthreads();
    }

    // epilogue (C layout: VGPR j -> row j / 8+j, col = lane&15)
#pragma unroll
    for (int mt = 0; mt < 4; ++mt) {
#pragma unroll
        for (int nt = 0; nt < 4; ++nt) {
            int colg = bn + wn * 64 + nt * 16 + l15;
            float bv = bias[colg];
#pragma unroll
            for (int j = 0; j < 8; ++j) {
                int rowg = bm + wm * 64 + mt * 16 + ((lane < 16) ? j : 8 + j);
                float v = acc[mt][nt][j] + bv;
                if (epilogue == EP_GELU) v = gelu_exact(v);
                if (epilogue == EP_RES)  v += Res[(size_t)rowg * N + colg];
                if (Cb) Cb[(size_t)rowg * N + colg] = (__bf16)v;
                else    Cf[(size_t)rowg * N + colg] = v;
            }
        }
    }
}

// ---------------------------------------------------------------------------
// LayerNorm over last dim (D=1024), one row per 256-thread block.
// Optional fused GELU and positional-embedding add.
// ---------------------------------------------------------------------------
__global__ __launch_bounds__(256)
void layernorm_k(const float* __restrict__ X, const float* __restrict__ g,
                 const float* __restrict__ bta, const float* __restrict__ pos,
                 float* __restrict__ Y, int Dn, int Tlen, int do_gelu)
{
    __shared__ float red[256];
    const int tid = threadIdx.x;
    const int row = blockIdx.x;
    const float* xr = X + (size_t)row * Dn;
    const int per = Dn >> 8;   // 4 for D=1024

    float local[4];
    float s = 0.f;
    for (int i = 0; i < per; ++i) { local[i] = xr[i * 256 + tid]; s += local[i]; }
    red[tid] = s; __syncthreads();
    for (int off = 128; off > 0; off >>= 1) {
        if (tid < off) red[tid] += red[tid + off];
        __syncthreads();
    }
    float mean = red[0] / (float)Dn;
    __syncthreads();

    float v = 0.f;
    for (int i = 0; i < per; ++i) { float d = local[i] - mean; v += d * d; }
    red[tid] = v; __syncthreads();
    for (int off = 128; off > 0; off >>= 1) {
        if (tid < off) red[tid] += red[tid + off];
        __syncthreads();
    }
    float rstd = rsqrtf(red[0] / (float)Dn + 1e-5f);

    const int t = row % Tlen;
    for (int i = 0; i < per; ++i) {
        int c = i * 256 + tid;
        float o = (local[i] - mean) * rstd * g[c] + bta[c];
        if (do_gelu) o = gelu_exact(o);
        if (pos) o += pos[(size_t)t * Dn + c];
        Y[(size_t)row * Dn + c] = o;
    }
}

// ---------------------------------------------------------------------------
// Causal multi-head attention, flash-style. One wave per 16-query tile of
// one (batch, head). Q/K/V bf16 [B,T,D] with head h at columns h*64..h*64+63.
// Keys streamed in chunks of 32; S = Q K^T and O += P V via WMMA.
// ---------------------------------------------------------------------------
__global__ __launch_bounds__(128)
void attn_wmma(const __bf16* __restrict__ Qm, const __bf16* __restrict__ Km,
               const __bf16* __restrict__ Vm, float* __restrict__ Y)
{
    __shared__ __bf16 Pb[4][16 * 32];   // per-wave P slab  [qrow][key]
    __shared__ __bf16 Vt[4][64 * 32];   // per-wave V^T slab [d][key]

    const int lane = threadIdx.x & 31;
    const int w    = threadIdx.x >> 5;
    const int bh   = blockIdx.y;
    const int bIdx = bh >> 4;          // H = 16
    const int h    = bh & 15;
    const int t0   = (blockIdx.x * 4 + w) * 16;
    const size_t rowBase = (size_t)bIdx * TSEQ;
    const int hd   = h * DHEAD;

    const int l15 = lane & 15;
    const int klo = (lane < 16) ? 0 : 8;
    const int ks  = (lane < 16) ? 0 : 16;

    // Q A-frags for dh slices [0,32) and [32,64)
    FragA qf[2];
    {
        const __bf16* qrow = Qm + (rowBase + t0 + l15) * DMODEL + hd;
#pragma unroll
        for (int kb = 0; kb < 2; ++kb) {
            qf[kb].h[0] = *(const v8bf*)(qrow + kb * 32 + ks);
            qf[kb].h[1] = *(const v8bf*)(qrow + kb * 32 + ks + 8);
        }
    }

    v8f zero = {};
    v8f oacc[4];
#pragma unroll
    for (int nt = 0; nt < 4; ++nt) oacc[nt] = zero;
    float m_i[8], l_i[8];
#pragma unroll
    for (int j = 0; j < 8; ++j) { m_i[j] = -3.0e38f; l_i[j] = 0.f; }

    const float scale = 0.125f;        // 1/sqrt(64)
    const int nch = (t0 + 16 + 31) >> 5;

    for (int c = 0; c < nch; ++c) {
        const int s0 = c * 32;

        // prefetch next chunk's K/V rows while computing this one
        if (c + 1 < nch) {
            __builtin_prefetch(Km + (rowBase + s0 + 32 + lane) * DMODEL + hd, 0, 0);
            __builtin_prefetch(Vm + (rowBase + s0 + 32 + lane) * DMODEL + hd, 0, 0);
        }

        // batch all K-frag loads (and V rows) before the WMMAs
        v16bf kf[2][2];   // [key-tile][kb]
#pragma unroll
        for (int kb = 0; kb < 2; ++kb) {
            kf[0][kb] = *(const v16bf*)(Km + (rowBase + s0 + l15) * DMODEL + hd + kb * 32 + ks);
            kf[1][kb] = *(const v16bf*)(Km + (rowBase + s0 + 16 + l15) * DMODEL + hd + kb * 32 + ks);
        }
        v8bf vv[8];
        {
            const __bf16* vrow = Vm + (rowBase + s0 + lane) * DMODEL + hd;
#pragma unroll
            for (int dc = 0; dc < 8; ++dc) vv[dc] = *(const v8bf*)(vrow + dc * 8);
        }

        // S tiles: two 16-key tiles, each contracting dh=64 (2 WMMA each)
        v8f sA = zero, sB = zero;
#pragma unroll
        for (int kb = 0; kb < 2; ++kb) {
            sA = __builtin_amdgcn_wmma_f32_16x16x32_bf16(false, qf[kb].v, false, kf[0][kb],
                                                         (short)0, sA, false, false);
            sB = __builtin_amdgcn_wmma_f32_16x16x32_bf16(false, qf[kb].v, false, kf[1][kb],
                                                         (short)0, sB, false, false);
        }

        // stage V chunk transposed: Vt[d][key], one key-row per lane
#pragma unroll
        for (int dc = 0; dc < 8; ++dc)
#pragma unroll
            for (int e = 0; e < 8; ++e)
                Vt[w][(dc * 8 + e) * 32 + lane] = vv[dc][e];

        const bool domask = (s0 + 31 > t0);

        // online softmax per row (rows striped across the two 16-lane halves)
#pragma unroll
        for (int j = 0; j < 8; ++j) {
            const int m  = (lane < 16) ? j : j + 8;
            const int tq = t0 + m;
            float a0 = sA[j] * scale;
            float a1 = sB[j] * scale;
            if (domask) {
                if (s0 + l15 > tq)       a0 = -3.0e38f;
                if (s0 + 16 + l15 > tq)  a1 = -3.0e38f;
            }
            float mx = fmaxf(a0, a1);
            mx = fmaxf(mx, __shfl_xor(mx, 1, 32));
            mx = fmaxf(mx, __shfl_xor(mx, 2, 32));
            mx = fmaxf(mx, __shfl_xor(mx, 4, 32));
            mx = fmaxf(mx, __shfl_xor(mx, 8, 32));
            const float nm    = fmaxf(m_i[j], mx);
            const float alpha = __expf(m_i[j] - nm);
            const float p0 = __expf(a0 - nm);
            const float p1 = __expf(a1 - nm);
            float rs = p0 + p1;
            rs += __shfl_xor(rs, 1, 32);
            rs += __shfl_xor(rs, 2, 32);
            rs += __shfl_xor(rs, 4, 32);
            rs += __shfl_xor(rs, 8, 32);
            l_i[j] = l_i[j] * alpha + rs;
            m_i[j] = nm;
#pragma unroll
            for (int nt = 0; nt < 4; ++nt) oacc[nt][j] *= alpha;
            Pb[w][m * 32 + l15]      = (__bf16)p0;
            Pb[w][m * 32 + 16 + l15] = (__bf16)p1;
        }

        // O += P V : P as A-frag (K=32 keys), V as 4 B-frags (dh tiles of 16)
        FragA pf;
        pf.h[0] = *(const v8bf*)&Pb[w][l15 * 32 + ks];
        pf.h[1] = *(const v8bf*)&Pb[w][l15 * 32 + ks + 8];
#pragma unroll
        for (int nt = 0; nt < 4; ++nt) {
            v16bf vf = *(const v16bf*)&Vt[w][(nt * 16 + l15) * 32 + ks];
            oacc[nt] = __builtin_amdgcn_wmma_f32_16x16x32_bf16(
                false, pf.v, false, vf, (short)0, oacc[nt], false, false);
        }
    }

    // normalize and store fp32 output
#pragma unroll
    for (int nt = 0; nt < 4; ++nt) {
#pragma unroll
        for (int j = 0; j < 8; ++j) {
            const int m = (lane < 16) ? j : j + 8;
            float o = oacc[nt][j] / l_i[j];
            Y[(rowBase + t0 + m) * DMODEL + hd + nt * 16 + l15] = o;
        }
    }
}

// ---------------------------------------------------------------------------
extern "C" void kernel_launch(void* const* d_in, const int* in_sizes, int n_in,
                              void* d_out, int out_size, void* d_ws, size_t ws_size,
                              hipStream_t stream)
{
    const float* obs    = (const float*)d_in[0];
    const float* enc_w1 = (const float*)d_in[1];
    const float* enc_b1 = (const float*)d_in[2];
    const float* enc_g1 = (const float*)d_in[3];
    const float* enc_be1= (const float*)d_in[4];
    const float* enc_w2 = (const float*)d_in[5];
    const float* enc_b2 = (const float*)d_in[6];
    const float* enc_w3 = (const float*)d_in[7];
    const float* enc_b3 = (const float*)d_in[8];
    const float* enc_g2 = (const float*)d_in[9];
    const float* enc_be2= (const float*)d_in[10];
    const float* pos_emb= (const float*)d_in[11];
    const float* ln1_g  = (const float*)d_in[12];
    const float* ln1_b  = (const float*)d_in[13];
    const float* wq     = (const float*)d_in[14];
    const float* bq     = (const float*)d_in[15];
    const float* wk     = (const float*)d_in[16];
    const float* bk     = (const float*)d_in[17];
    const float* wv     = (const float*)d_in[18];
    const float* bv     = (const float*)d_in[19];
    const float* wo     = (const float*)d_in[20];
    const float* bo     = (const float*)d_in[21];
    const float* ln2_g  = (const float*)d_in[22];
    const float* ln2_b  = (const float*)d_in[23];
    const float* mlp_w1 = (const float*)d_in[24];
    const float* mlp_b1 = (const float*)d_in[25];
    const float* mlp_w2 = (const float*)d_in[26];
    const float* mlp_b2 = (const float*)d_in[27];
    const float* lnf_g  = (const float*)d_in[28];
    const float* lnf_b  = (const float*)d_in[29];
    const float* dec_w  = (const float*)d_in[30];
    const float* dec_b  = (const float*)d_in[31];

    // workspace carve-up
    char* ws = (char*)d_ws;
    size_t off = 0;
    float* x   = (float*)(ws + off); off += (size_t)MROWS * DMODEL * sizeof(float);
    float* t0  = (float*)(ws + off); off += (size_t)MROWS * DMODEL * sizeof(float);
    float* t1  = (float*)(ws + off); off += (size_t)MROWS * DMODEL * sizeof(float);
    float* yb  = (float*)(ws + off); off += (size_t)MROWS * DMODEL * sizeof(float);
    float* ff  = (float*)(ws + off); off += (size_t)MROWS * FFDIM  * sizeof(float);
    __bf16* qb = (__bf16*)(ws + off); off += (size_t)MROWS * DMODEL * sizeof(__bf16);
    __bf16* kb = (__bf16*)(ws + off); off += (size_t)MROWS * DMODEL * sizeof(__bf16);
    __bf16* vb = (__bf16*)(ws + off); off += (size_t)MROWS * DMODEL * sizeof(__bf16);

    const dim3 gblk(256);

    auto gemmF = [&](const float* A, const float* W, const float* b,
                     float* C, const float* Res, int M, int K, int N, int ep) {
        dim3 grid(N / 256, M / 128);
        gemm_wmma_bf16<<<grid, gblk, 0, stream>>>(A, W, b, Res, C, (__bf16*)nullptr,
                                                  M, N, K, ep);
    };
    auto gemmB = [&](const float* A, const float* W, const float* b,
                     __bf16* C, int M, int K, int N) {
        dim3 grid(N / 256, M / 128);
        gemm_wmma_bf16<<<grid, gblk, 0, stream>>>(A, W, b, nullptr, (float*)nullptr, C,
                                                  M, N, K, EP_NONE);
    };
    auto lnorm = [&](const float* X, const float* g, const float* b,
                     const float* pos, float* Yd, int do_gelu) {
        layernorm_k<<<dim3(MROWS), dim3(256), 0, stream>>>(X, g, b, pos, Yd,
                                                           DMODEL, TSEQ, do_gelu);
    };

    // ---- encoder MLP ----
    gemmF(obs, enc_w1, enc_b1, t0, nullptr, MROWS, ODIM, DMODEL, EP_NONE);
    lnorm(t0, enc_g1, enc_be1, nullptr, t1, /*gelu=*/1);
    gemmF(t1, enc_w2, enc_b2, t0, nullptr, MROWS, DMODEL, DMODEL, EP_GELU);
    gemmF(t0, enc_w3, enc_b3, t1, nullptr, MROWS, DMODEL, DMODEL, EP_NONE);
    lnorm(t1, enc_g2, enc_be2, pos_emb, x, /*gelu=*/0);   // fused + pos_emb

    // ---- transformer layers ----
    for (int l = 0; l < NLAYER; ++l) {
        const size_t wOff = (size_t)l * DMODEL * DMODEL;
        const size_t bOff = (size_t)l * DMODEL;
        lnorm(x, ln1_g + bOff, ln1_b + bOff, nullptr, t0, 0);
        gemmB(t0, wq + wOff, bq + bOff, qb, MROWS, DMODEL, DMODEL);
        gemmB(t0, wk + wOff, bk + bOff, kb, MROWS, DMODEL, DMODEL);
        gemmB(t0, wv + wOff, bv + bOff, vb, MROWS, DMODEL, DMODEL);
        attn_wmma<<<dim3(TSEQ / 64, BB * NHEAD), dim3(128), 0, stream>>>(qb, kb, vb, yb);
        gemmF(yb, wo + wOff, bo + bOff, x, x, MROWS, DMODEL, DMODEL, EP_RES);
        lnorm(x, ln2_g + bOff, ln2_b + bOff, nullptr, t0, 0);
        gemmF(t0, mlp_w1 + (size_t)l * DMODEL * FFDIM, mlp_b1 + (size_t)l * FFDIM,
              ff, nullptr, MROWS, DMODEL, FFDIM, EP_GELU);
        gemmF(ff, mlp_w2 + (size_t)l * FFDIM * DMODEL, mlp_b2 + bOff,
              x, x, MROWS, FFDIM, DMODEL, EP_RES);
    }

    // ---- final LN (first output) + decoder (second output) ----
    float* out_x   = (float*)d_out;
    float* out_obs = out_x + (size_t)MROWS * DMODEL;
    lnorm(x, lnf_g, lnf_b, nullptr, out_x, 0);
    gemmF(out_x, dec_w, dec_b, out_obs, nullptr, MROWS, DMODEL, ODIM, EP_NONE);
}
